// GATLayer_76390288326932
// MI455X (gfx1250) — compile-verified, matrix operand order checked by reading, not run on hardware
//
#include <hip/hip_runtime.h>
#include <hip/hip_bf16.h>

typedef __attribute__((ext_vector_type(16))) __bf16 v16bf;
typedef __attribute__((ext_vector_type(8)))  float  v8f;

// ---------------------------------------------------------------------------
// Fragment loader for V_WMMA_F32_16X16X32_BF16.
// 16-bit A (16x32) layout: lane L = row (L&15), holds K = h*8..h*8+7 in
// VGPR0-3 and K = h*8+16..h*8+23 in VGPR4-7, h = L>>4. B fragments (32x16)
// use the same pattern with lane = column, so any [rows x K] row-major matrix
// is fetched with two 16-byte loads per lane.
// ---------------------------------------------------------------------------
__device__ __forceinline__ v16bf load_frag(const __bf16* __restrict__ p, int ld,
                                           int rowbase, int kbase, int lane) {
  int r = lane & 15;
  int h = (lane >> 4) << 3;  // 0 or 8
  const __bf16* q = p + (size_t)(rowbase + r) * ld + kbase + h;
  union { uint4 u[2]; v16bf f; } tmp;
  tmp.u[0] = *(const uint4*)(q);
  tmp.u[1] = *(const uint4*)(q + 16);
  return tmp.f;
}

__device__ __forceinline__ v8f wmma_bf16(v16bf a, v16bf b, v8f c) {
  return __builtin_amdgcn_wmma_f32_16x16x32_bf16(false, a, false, b,
                                                 (short)0, c, false, false);
}

// ---------------------------------------------------------------------------
// f32 -> bf16 conversion
// ---------------------------------------------------------------------------
__global__ __launch_bounds__(256) void cvt_kernel(const float* __restrict__ src,
                                                  __bf16* __restrict__ dst, int n) {
  int i = blockIdx.x * 256 + threadIdx.x;
  if (i < n) dst[i] = (__bf16)src[i];
}

// ---------------------------------------------------------------------------
// Pack adjacency (int32 0/1, [B*N*N]) into a bitmask via wave ballot:
// bm[i>>5] bit (i&31) = adj[i] != 0.  8.4M ints -> 1 MB of mask bits.
// ---------------------------------------------------------------------------
__global__ __launch_bounds__(256) void maskpack_kernel(const int* __restrict__ adj,
                                                       unsigned* __restrict__ bm,
                                                       int n) {
  int i = blockIdx.x * 256 + threadIdx.x;
  unsigned long long ball = __ballot((i < n) && (adj[i] != 0));
  if ((threadIdx.x & 31) == 0) bm[i >> 5] = (unsigned)ball;
}

// ---------------------------------------------------------------------------
// GEMM: out[m,n] = sum_k A[m,k] * W[n,k] + bias[n]   (M=4096, K=512, Nout=512)
// mode 0: bf16 out, per-head layout   [BH, N, 64]      (Q, K)
// mode 1: bf16 out, transposed heads  [BH, 64, N]      (V)
// mode 2: f32 out + residual x        [M, 512]         (output projection)
// wave -> 32x32 tile; block = 8 waves covering 32 x 256.
// ---------------------------------------------------------------------------
__global__ __launch_bounds__(256) void gemm_kernel(
    const __bf16* __restrict__ A, const __bf16* __restrict__ W,
    const float* __restrict__ bias, const float* __restrict__ resid,
    __bf16* __restrict__ out_bf, float* __restrict__ out_f, int K, int mode) {
  int lane  = threadIdx.x & 31;
  int w     = threadIdx.x >> 5;
  int mbase = blockIdx.x * 32;
  int nbase = blockIdx.y * 256 + w * 32;

  v8f acc00 = {}, acc01 = {}, acc10 = {}, acc11 = {};
  for (int k = 0; k < K; k += 32) {
    v16bf a0 = load_frag(A, K, mbase,      k, lane);
    v16bf a1 = load_frag(A, K, mbase + 16, k, lane);
    v16bf b0 = load_frag(W, K, nbase,      k, lane);
    v16bf b1 = load_frag(W, K, nbase + 16, k, lane);
    acc00 = wmma_bf16(a0, b0, acc00);
    acc01 = wmma_bf16(a0, b1, acc01);
    acc10 = wmma_bf16(a1, b0, acc10);
    acc11 = wmma_bf16(a1, b1, acc11);
  }

  // C/D layout: lane c (<16) holds rows 0..7 of column c, lane 16+c rows 8..15
  int col  = lane & 15;
  int rtop = (lane >> 4) << 3;
  v8f accs[2][2] = {{acc00, acc01}, {acc10, acc11}};
#pragma unroll
  for (int i = 0; i < 2; ++i)
#pragma unroll
    for (int j = 0; j < 2; ++j) {
      int n = nbase + j * 16 + col;
      float bv = bias[n];
#pragma unroll
      for (int r = 0; r < 8; ++r) {
        int m = mbase + i * 16 + rtop + r;
        float val = accs[i][j][r] + bv;
        if (mode == 2) {
          out_f[(size_t)m * 512 + n] = val + resid[(size_t)m * 512 + n];
        } else {
          int bb = m >> 11, t = m & 2047;   // batch, token
          int h  = n >> 6,  hd = n & 63;    // head, head-dim
          size_t idx = (mode == 0)
              ? ((size_t)((bb * 8 + h) * 2048 + t) * 64 + hd)
              : ((size_t)((bb * 8 + h) * 64 + hd) * 2048 + t);
          out_bf[idx] = (__bf16)val;
        }
      }
    }
}

// ---------------------------------------------------------------------------
// Flash attention with packed adjacency mask. One wave per 16-row q-tile,
// 64 keys per iteration (16 WMMAs / iteration, softmax bookkeeping amortized).
// qb/kb: [BH,2048,64] bf16, vtb: [BH,64,2048] bf16, bm: packed bits,
// ctx: [B*2048, 512] bf16.
// ---------------------------------------------------------------------------
__global__ __launch_bounds__(128) void attn_kernel(
    const __bf16* __restrict__ qb, const __bf16* __restrict__ kb,
    const __bf16* __restrict__ vtb, const unsigned* __restrict__ bm,
    __bf16* __restrict__ ctx) {
  __shared__ __bf16 lds_p[4][16 * 64];

  int lane = threadIdx.x & 31;
  int w    = threadIdx.x >> 5;
  int tile = blockIdx.x * 4 + w;   // 0..2047
  int qt   = tile & 127;
  int bh   = tile >> 7;
  int b    = bh >> 3;
  int h    = bh & 7;

  const __bf16* qB = qb  + (size_t)bh * 2048 * 64;
  const __bf16* kB = kb  + (size_t)bh * 2048 * 64;
  const __bf16* vB = vtb + (size_t)bh * 64 * 2048;

  v16bf aq0 = load_frag(qB, 64, qt * 16, 0,  lane);
  v16bf aq1 = load_frag(qB, 64, qt * 16, 32, lane);

  v8f o[4] = {};
  float mrun[8], lrun[8];
#pragma unroll
  for (int r = 0; r < 8; ++r) { mrun[r] = -1e30f; lrun[r] = 0.f; }

  int col  = lane & 15;
  int rtop = (lane >> 4) << 3;
  // packed mask: 64 words (2048 bits) per q-row
  const unsigned* bmRow = bm + ((size_t)b * 2048 + qt * 16 + rtop) * 64;
  const float scale = 0.125f;  // 1/sqrt(64)

  for (int k0 = 0; k0 < 2048; k0 += 64) {
    // S = Q (16x64) . K^T -> four 16x16 subtiles, 2 WMMAs each over hd
    v8f s[4];
#pragma unroll
    for (int j = 0; j < 4; ++j) {
      v8f t = {};
      t = wmma_bf16(aq0, load_frag(kB, 64, k0 + j * 16, 0,  lane), t);
      t = wmma_bf16(aq1, load_frag(kB, 64, k0 + j * 16, 32, lane), t);
      s[j] = t;
    }

    // mask + scale: one 8-byte bitmask load per row covers all 64 keys
    float p[4][8], cm[8];
#pragma unroll
    for (int r = 0; r < 8; ++r) {
      uint2 mw = *(const uint2*)(bmRow + (size_t)r * 64 + (k0 >> 5));
#pragma unroll
      for (int j = 0; j < 4; ++j) {
        unsigned word = (j < 2) ? mw.x : mw.y;
        unsigned bit  = (word >> (((j & 1) << 4) + col)) & 1u;
        p[j][r] = bit ? s[j][r] * scale : -1e30f;
      }
      cm[r] = fmaxf(fmaxf(p[0][r], p[1][r]), fmaxf(p[2][r], p[3][r]));
    }
    // row max across 16 lanes (xor 1/2/4/8 stays inside each half)
#pragma unroll
    for (int d = 1; d < 16; d <<= 1)
#pragma unroll
      for (int r = 0; r < 8; ++r)
        cm[r] = fmaxf(cm[r], __shfl_xor(cm[r], d, 32));

    float alpha[8], rs[8];
#pragma unroll
    for (int r = 0; r < 8; ++r) {
      float mn = fmaxf(mrun[r], cm[r]);
      alpha[r] = __expf(mrun[r] - mn);
      rs[r] = 0.f;
#pragma unroll
      for (int j = 0; j < 4; ++j) {
        p[j][r] = (p[j][r] > -1e29f) ? __expf(p[j][r] - mn) : 0.f;
        rs[r] += p[j][r];
      }
      mrun[r] = mn;
    }
#pragma unroll
    for (int d = 1; d < 16; d <<= 1)
#pragma unroll
      for (int r = 0; r < 8; ++r)
        rs[r] += __shfl_xor(rs[r], d, 32);
#pragma unroll
    for (int r = 0; r < 8; ++r)
      lrun[r] = lrun[r] * alpha[r] + rs[r];
#pragma unroll
    for (int t = 0; t < 4; ++t)
#pragma unroll
      for (int r = 0; r < 8; ++r)
        o[t][r] *= alpha[r];

    // stage P (16 x 64, bf16) in LDS: D-layout -> row-major
    __bf16* lp = lds_p[w];
#pragma unroll
    for (int j = 0; j < 4; ++j)
#pragma unroll
      for (int r = 0; r < 8; ++r)
        lp[(rtop + r) * 64 + j * 16 + col] = (__bf16)p[j][r];
    asm volatile("s_wait_dscnt 0" ::: "memory");  // cross-lane LDS RAW

    // O += P (16x64) . V (64x64): reload P as two A fragments,
    // transposed-V layout gives contiguous b128 B fragments.
    v16bf pa0 = load_frag(lp, 64, 0, 0,  lane);
    v16bf pa1 = load_frag(lp, 64, 0, 32, lane);
#pragma unroll
    for (int t = 0; t < 4; ++t) {
      o[t] = wmma_bf16(pa0, load_frag(vB, 2048, t * 16, k0,      lane), o[t]);
      o[t] = wmma_bf16(pa1, load_frag(vB, 2048, t * 16, k0 + 32, lane), o[t]);
    }
    asm volatile("" ::: "memory");  // keep next iter's LDS stores after the reads
  }

  // normalize and emit ctx[B*2048, 512] (bf16) for the output projection
  float inv[8];
#pragma unroll
  for (int r = 0; r < 8; ++r) inv[r] = (lrun[r] > 0.f) ? 1.0f / lrun[r] : 0.f;
#pragma unroll
  for (int t = 0; t < 4; ++t)
#pragma unroll
    for (int r = 0; r < 8; ++r) {
      int tok = qt * 16 + rtop + r;
      ctx[(size_t)(b * 2048 + tok) * 512 + h * 64 + t * 16 + col] =
          (__bf16)(o[t][r] * inv[r]);
    }
}

// ---------------------------------------------------------------------------
// LayerNorm over rows of 512 (f32 in, f32 out)
// ---------------------------------------------------------------------------
__global__ __launch_bounds__(256) void ln_kernel(const float* __restrict__ y,
                                                 const float* __restrict__ gamma,
                                                 const float* __restrict__ beta,
                                                 float* __restrict__ out) {
  int row = blockIdx.x;
  int tid = threadIdx.x;
  const float* yr = y + (size_t)row * 512;
  float v0 = yr[tid], v1 = yr[tid + 256];
  float s = v0 + v1, q = v0 * v0 + v1 * v1;
#pragma unroll
  for (int d = 1; d < 32; d <<= 1) {
    s += __shfl_xor(s, d, 32);
    q += __shfl_xor(q, d, 32);
  }
  __shared__ float ss[8], sq[8];
  int wv = tid >> 5, lane = tid & 31;
  if (lane == 0) { ss[wv] = s; sq[wv] = q; }
  __syncthreads();
  float ts = 0.f, tq = 0.f;
#pragma unroll
  for (int i = 0; i < 8; ++i) { ts += ss[i]; tq += sq[i]; }
  float mu   = ts * (1.0f / 512.0f);
  float var  = tq * (1.0f / 512.0f) - mu * mu;
  float rinv = rsqrtf(var + 1e-5f);
  out[(size_t)row * 512 + tid]       = (v0 - mu) * rinv * gamma[tid] + beta[tid];
  out[(size_t)row * 512 + tid + 256] = (v1 - mu) * rinv * gamma[tid + 256] + beta[tid + 256];
}

// ---------------------------------------------------------------------------
extern "C" void kernel_launch(void* const* d_in, const int* in_sizes, int n_in,
                              void* d_out, int out_size, void* d_ws, size_t ws_size,
                              hipStream_t stream) {
  const float* x     = (const float*)d_in[0];
  const int*   adj   = (const int*)d_in[1];
  const float* Wq    = (const float*)d_in[2];
  const float* bq    = (const float*)d_in[3];
  const float* Wk    = (const float*)d_in[4];
  const float* bk    = (const float*)d_in[5];
  const float* Wv    = (const float*)d_in[6];
  const float* bv    = (const float*)d_in[7];
  const float* Wo    = (const float*)d_in[8];
  const float* bo    = (const float*)d_in[9];
  const float* gamma = (const float*)d_in[10];
  const float* beta  = (const float*)d_in[11];
  float* out = (float*)d_out;

  char* ws = (char*)d_ws;
  const size_t MB = 1024 * 1024;
  __bf16*   xb   = (__bf16*)(ws + 0);                 // 4 MiB   x bf16
  __bf16*   wqb  = (__bf16*)(ws + 4 * MB);            // 512 KiB
  __bf16*   wkb  = (__bf16*)(ws + 4 * MB + 512 * 1024);
  __bf16*   wvb  = (__bf16*)(ws + 5 * MB);
  __bf16*   wob  = (__bf16*)(ws + 5 * MB + 512 * 1024);
  __bf16*   qb   = (__bf16*)(ws + 6 * MB);            // 4 MiB  [BH,N,64]
  __bf16*   kb   = (__bf16*)(ws + 10 * MB);           // 4 MiB  [BH,N,64]
  __bf16*   vtb  = (__bf16*)(ws + 14 * MB);           // 4 MiB  [BH,64,N]
  __bf16*   ctx  = (__bf16*)(ws + 18 * MB);           // 4 MiB  [B*N,512]
  float*    ypre = (float*)(ws + 22 * MB);            // 8 MiB  [B*N,512]
  unsigned* bmw  = (unsigned*)(ws + 30 * MB);         // 1 MiB  packed adjacency

  cvt_kernel<<<8192, 256, 0, stream>>>(x,  xb,  4096 * 512);
  cvt_kernel<<<1024, 256, 0, stream>>>(Wq, wqb, 512 * 512);
  cvt_kernel<<<1024, 256, 0, stream>>>(Wk, wkb, 512 * 512);
  cvt_kernel<<<1024, 256, 0, stream>>>(Wv, wvb, 512 * 512);
  cvt_kernel<<<1024, 256, 0, stream>>>(Wo, wob, 512 * 512);
  maskpack_kernel<<<32768, 256, 0, stream>>>(adj, bmw, 2 * 2048 * 2048);

  dim3 gg(128, 2), gb(256);
  gemm_kernel<<<gg, gb, 0, stream>>>(xb, wqb, bq, x, qb,  ypre, 512, 0);
  gemm_kernel<<<gg, gb, 0, stream>>>(xb, wkb, bk, x, kb,  ypre, 512, 0);
  gemm_kernel<<<gg, gb, 0, stream>>>(xb, wvb, bv, x, vtb, ypre, 512, 1);

  attn_kernel<<<512, 128, 0, stream>>>(qb, kb, vtb, bmw, ctx);

  gemm_kernel<<<gg, gb, 0, stream>>>(ctx, wob, bo, x, qb, ypre, 512, 2);

  ln_kernel<<<4096, 256, 0, stream>>>(ypre, gamma, beta, out);
}
